// RelPartialLearnableMultiHeadAttn_8710193676750
// MI455X (gfx1250) — matrix operand order repaired and checked by reading, not compiled, not run on hardware
//
#include <hip/hip_runtime.h>
#include <math.h>

// ---------------------------------------------------------------------------
// Types for CDNA5 WMMA (wave32). bf16 in, f32 accumulate.
// ---------------------------------------------------------------------------
typedef __attribute__((ext_vector_type(16))) __bf16 v16bf;
typedef __attribute__((ext_vector_type(8)))  float  v8f;
typedef int v4i_t __attribute__((vector_size(16)));  // async-LDS builtin pointee

#if defined(__HIP_DEVICE_COMPILE__) &&                                      \
    __has_builtin(__builtin_amdgcn_global_load_async_to_lds_b128) &&        \
    __has_builtin(__builtin_amdgcn_s_wait_asynccnt)
#define ASYNC_LDS 1
#else
#define ASYNC_LDS 0
#endif

__device__ __forceinline__ v8f wmma_bf16(v16bf a, v16bf b, v8f c) {
  // (neg_a, A, neg_b, B, c_mod, C, reuse_a, reuse_b)
  return __builtin_amdgcn_wmma_f32_16x16x32_bf16(false, a, false, b, (short)0, c,
                                                 false, false);
}

// A fragment (16x32, 16-bit): lane holds row M = lane%16.
// elems e<8 -> K = 8h+e ; e>=8 -> K = 16+8h+(e-8)   (ISA 7.12.2 table)
__device__ __forceinline__ v16bf a_frag(const __bf16* rowbase, int h) {
  v16bf v;
  const __bf16* p = rowbase + 8 * h;
#pragma unroll
  for (int e = 0; e < 8; ++e) { v[e] = p[e]; v[8 + e] = p[16 + e]; }
  return v;
}

// B fragment (32x16, 16-bit) where the source S is stored row-major with rows
// indexed by the output column (i.e. B = S^T): lane holds column n = lane%16;
// elems e<8 -> K = 8h+e ; e>=8 -> K = 16+8h+(e-8) (same as A fragment).
__device__ __forceinline__ v16bf bT_frag(const __bf16* rowbase, int h) {
  v16bf v;
  const __bf16* p = rowbase + 8 * h;
#pragma unroll
  for (int e = 0; e < 8; ++e) { v[e] = p[e]; v[8 + e] = p[16 + e]; }
  return v;
}

// ---------------------------------------------------------------------------
// Sizes
// ---------------------------------------------------------------------------
#define QLEN  1024
#define BSZ   4
#define DMOD  512
#define NHEAD 8
#define DHEAD 64
#define MROWS (QLEN * BSZ)          // 4096
#define HCOLS (3 * NHEAD * DHEAD)   // 1536
#define LDH   (BSZ * HCOLS)         // 6144 : row stride over i for fixed b

// ---------------------------------------------------------------------------
// Prep 1: f32 -> bf16 convert (row-major preserved), vectorized x4
// ---------------------------------------------------------------------------
__launch_bounds__(256)
__global__ void convert_bf16(const float* __restrict__ in,
                             __bf16* __restrict__ out, int n4) {
  int i = blockIdx.x * 256 + threadIdx.x;
  if (i >= n4) return;
  float4 v = ((const float4*)in)[i];
  __bf16* o = out + (size_t)i * 4;
  o[0] = (__bf16)v.x; o[1] = (__bf16)v.y; o[2] = (__bf16)v.z; o[3] = (__bf16)v.w;
}

// ---------------------------------------------------------------------------
// Prep 2: out[N][K] bf16 = transpose(in[K][N] f32), LDS-tiled, coalesced
// ---------------------------------------------------------------------------
__launch_bounds__(256)
__global__ void transpose_to_bf16(const float* __restrict__ in,
                                  __bf16* __restrict__ out, int K, int N) {
  __shared__ float tile[32][33];
  int kt = blockIdx.x * 32, nt = blockIdx.y * 32;
  int tx = threadIdx.x & 31, ty = threadIdx.x >> 5;  // 32 x 8
#pragma unroll
  for (int i = 0; i < 4; ++i)
    tile[ty + 8 * i][tx] = in[(size_t)(kt + ty + 8 * i) * N + nt + tx];
  __syncthreads();
#pragma unroll
  for (int i = 0; i < 4; ++i)
    out[(size_t)(nt + ty + 8 * i) * K + kt + tx] = (__bf16)tile[tx][ty + 8 * i];
}

// ---------------------------------------------------------------------------
// Prep 3: repack V per head: vt[b][n][d][j] = heads[(j*4+b)*1536+1024+n*64+d]
// Global->LDS staging uses the gfx1250 async-to-LDS path when available
// (tracked with ASYNCcnt, completed with s_wait_asynccnt).
// ---------------------------------------------------------------------------
__launch_bounds__(256)
__global__ void repack_v(const __bf16* __restrict__ h, __bf16* __restrict__ vt) {
  int jt = blockIdx.x * 64;
  int b = blockIdx.y >> 3, n = blockIdx.y & 7;
  // 72-elem rows: 144B stride = 9*16B -> every 16B async chunk stays aligned,
  // and the padding breaks up bank conflicts on the transposed read.
  __shared__ __align__(16) __bf16 t[64][72];
  int tx = threadIdx.x & 63, ty = threadIdx.x >> 6;  // 64 x 4
  const __bf16* src = h + b * HCOLS + 2 * NHEAD * DHEAD + n * DHEAD;
#if ASYNC_LDS
  {
    // 64x64 bf16 tile = 512 chunks of 16B; 2 chunks per thread, direct to LDS.
#pragma unroll
    for (int it = 0; it < 2; ++it) {
      int c = threadIdx.x + 256 * it;
      int row = c >> 3, ch = c & 7;
      const __bf16* g = src + (size_t)(jt + row) * LDH + ch * 8;
      __builtin_amdgcn_global_load_async_to_lds_b128(
          (v4i_t*)g, (v4i_t*)&t[row][ch * 8], 0, 0);
    }
    __builtin_amdgcn_s_wait_asynccnt(0);
  }
#else
#pragma unroll
  for (int i = 0; i < 16; ++i)
    t[ty + 4 * i][tx] = src[(size_t)(jt + ty + 4 * i) * LDH + tx];
#endif
  __syncthreads();
  __bf16* dst = vt + (size_t)(b * NHEAD + n) * DHEAD * QLEN;
#pragma unroll
  for (int i = 0; i < 16; ++i)
    dst[(size_t)(ty + 4 * i) * QLEN + jt + tx] = t[tx][ty + 4 * i];
}

// ---------------------------------------------------------------------------
// GEMM: C_bf16[M,N] = A_bf16[M,K] @ Bt_bf16[N,K]^T, 32x64 tile per wave
// ---------------------------------------------------------------------------
__launch_bounds__(128)
__global__ void gemm_bf16(const __bf16* __restrict__ A,
                          const __bf16* __restrict__ Bt,
                          __bf16* __restrict__ C, int M, int N, int K) {
  int wid = blockIdx.x * 4 + (threadIdx.x >> 5);
  int ntile = N >> 6;
  if (wid >= (M >> 5) * ntile) return;
  int m0 = (wid / ntile) << 5, n0 = (wid % ntile) << 6;
  int lane = threadIdx.x & 31, nn = lane & 15, h = lane >> 4;
  v8f acc[2][4] = {};
  for (int k0 = 0; k0 < K; k0 += 32) {
    __builtin_prefetch(A + (size_t)(m0 + nn) * K + k0 + 64, 0, 1);
    __builtin_prefetch(Bt + (size_t)(n0 + nn) * K + k0 + 64, 0, 1);
    v16bf a0 = a_frag(A + (size_t)(m0 + nn) * K + k0, h);
    v16bf a1 = a_frag(A + (size_t)(m0 + 16 + nn) * K + k0, h);
#pragma unroll
    for (int t = 0; t < 4; ++t) {
      v16bf bb = bT_frag(Bt + (size_t)(n0 + 16 * t + nn) * K + k0, h);
      acc[0][t] = wmma_bf16(a0, bb, acc[0][t]);
      acc[1][t] = wmma_bf16(a1, bb, acc[1][t]);
    }
  }
#pragma unroll
  for (int mi = 0; mi < 2; ++mi) {
    __bf16* pc = C + (size_t)(m0 + 16 * mi + 8 * h) * N + n0 + nn;
#pragma unroll
    for (int t = 0; t < 4; ++t)
#pragma unroll
      for (int rr = 0; rr < 8; ++rr)
        pc[(size_t)rr * N + 16 * t] = (__bf16)acc[mi][t][rr];
  }
}

// ---------------------------------------------------------------------------
// GEMM + residual: Cout_f32[M,N] = A_bf16[M,K] @ Bt^T + resid_f32[M,N]
// ---------------------------------------------------------------------------
__launch_bounds__(128)
__global__ void gemm_resid(const __bf16* __restrict__ A,
                           const __bf16* __restrict__ Bt,
                           const float* __restrict__ resid,
                           float* __restrict__ Cout, int M, int N, int K) {
  int wid = blockIdx.x * 4 + (threadIdx.x >> 5);
  int ntile = N >> 6;
  if (wid >= (M >> 5) * ntile) return;
  int m0 = (wid / ntile) << 5, n0 = (wid % ntile) << 6;
  int lane = threadIdx.x & 31, nn = lane & 15, h = lane >> 4;
  v8f acc[2][4] = {};
  for (int k0 = 0; k0 < K; k0 += 32) {
    __builtin_prefetch(A + (size_t)(m0 + nn) * K + k0 + 64, 0, 1);
    __builtin_prefetch(Bt + (size_t)(n0 + nn) * K + k0 + 64, 0, 1);
    v16bf a0 = a_frag(A + (size_t)(m0 + nn) * K + k0, h);
    v16bf a1 = a_frag(A + (size_t)(m0 + 16 + nn) * K + k0, h);
#pragma unroll
    for (int t = 0; t < 4; ++t) {
      v16bf bb = bT_frag(Bt + (size_t)(n0 + 16 * t + nn) * K + k0, h);
      acc[0][t] = wmma_bf16(a0, bb, acc[0][t]);
      acc[1][t] = wmma_bf16(a1, bb, acc[1][t]);
    }
  }
#pragma unroll
  for (int mi = 0; mi < 2; ++mi) {
    size_t base = (size_t)(m0 + 16 * mi + 8 * h) * N + n0 + nn;
#pragma unroll
    for (int t = 0; t < 4; ++t)
#pragma unroll
      for (int rr = 0; rr < 8; ++rr)
        Cout[base + (size_t)rr * N + 16 * t] =
            acc[mi][t][rr] + resid[base + (size_t)rr * N + 16 * t];
  }
}

// ---------------------------------------------------------------------------
// Fused rel-attention (flash softmax): one (b,n) per blockIdx.y,
// 64 query rows per workgroup (4 waves x 16 rows). vec written bf16.
// ---------------------------------------------------------------------------
__launch_bounds__(128)
__global__ void attn_kernel(const __bf16* __restrict__ hr,
                            const __bf16* __restrict__ hp,
                            const __bf16* __restrict__ rh,
                            const __bf16* __restrict__ vtr,
                            const __bf16* __restrict__ vtp,
                            const float* __restrict__ rwb,
                            const float* __restrict__ rrb,
                            __bf16* __restrict__ vec_r,
                            __bf16* __restrict__ vec_p) {
  const int i0 = blockIdx.x * 64;
  const int b = blockIdx.y >> 3, n = blockIdx.y & 7;
  const int w = threadIdx.x >> 5;
  const int lane = threadIdx.x & 31;
  const int nn = lane & 15, h = lane >> 4;
  const int iw = i0 + 16 * w;

  const __bf16* qr = hr + b * HCOLS + n * DHEAD;
  const __bf16* kr = qr + NHEAD * DHEAD;
  const __bf16* qp = hp + b * HCOLS + n * DHEAD;
  const __bf16* kp = qp + NHEAD * DHEAD;
  const __bf16* rhb = rh + n * DHEAD;
  const __bf16* vr = vtr + (size_t)(b * NHEAD + n) * DHEAD * QLEN;  // [d][j]
  const __bf16* vp = vtp + (size_t)(b * NHEAD + n) * DHEAD * QLEN;

  __shared__ __align__(16) float  bd_r[4][2][16][32];
  __shared__ __align__(16) float  bd_p[4][2][16][32];
  __shared__ __align__(16) __bf16 pbuf[4][16][32];

  // ---- A fragments (loop-invariant): q + bias, bf16 ----
  v16bf arw_r[2], arw_p[2], arr_r[2], arr_p[2];
  {
    int row = iw + nn;
#pragma unroll
    for (int s = 0; s < 2; ++s) {
      int d0 = 32 * s + 8 * h;
      const __bf16* pr = qr + (size_t)row * LDH + d0;
      const __bf16* pp = qp + (size_t)row * LDH + d0;
      const float* bw = rwb + n * DHEAD + d0;
      const float* br = rrb + n * DHEAD + d0;
#pragma unroll
      for (int e = 0; e < 16; ++e) {
        int off = (e < 8) ? e : (8 + e);  // e>=8 -> 16+(e-8)
        float fr = (float)pr[off];
        float fp = (float)pp[off];
        arw_r[s][e] = (__bf16)(fr + bw[off]);
        arr_r[s][e] = (__bf16)(fr + br[off]);
        arw_p[s][e] = (__bf16)(fp + bw[off]);
        arr_p[s][e] = (__bf16)(fp + br[off]);
      }
    }
  }

  v8f out_r[4] = {}, out_p[4] = {};
  float mrow[8], lrow[8];
#pragma unroll
  for (int rr = 0; rr < 8; ++rr) { mrow[rr] = -INFINITY; lrow[rr] = 0.0f; }

  const float scale = 0.125f;      // 1/sqrt(64)
  const int njb = (i0 + 64) / 32;  // 32-key blocks (causal bound)

  for (int jb = 0; jb < njb; ++jb) {
    const int j0 = jb * 32;
    v8f ac_r[2], ac_pp[2], ac_rp[2];

    // ---- score WMMAs + stage BD tiles into LDS ----
#pragma unroll
    for (int sj = 0; sj < 2; ++sj) {
      const int jt = j0 + 16 * sj;
      v8f a0 = {}, a1 = {}, a2 = {};
      v8f b0 = {}, b1 = {}, b2 = {}, b3 = {};
      const int base = jt - iw + 1023;  // rel-shift column base
      int tn_ = base - 16 + nn; tn_ = tn_ < 0 ? 0 : (tn_ > 1023 ? 1023 : tn_);
      int tp_ = base + nn;      tp_ = tp_ < 0 ? 0 : (tp_ > 1023 ? 1023 : tp_);
#pragma unroll
      for (int s = 0; s < 2; ++s) {
        const int d0 = 32 * s;
        v16bf bkr = bT_frag(kr + (size_t)(jt + nn) * LDH + d0, h);
        v16bf bkp = bT_frag(kp + (size_t)(jt + nn) * LDH + d0, h);
        a0 = wmma_bf16(arw_r[s], bkr, a0);
        a1 = wmma_bf16(arw_p[s], bkp, a1);
        a2 = wmma_bf16(arw_r[s], bkp, a2);
        v16bf brn = bT_frag(rhb + (size_t)tn_ * DMOD + d0, h);
        v16bf brp = bT_frag(rhb + (size_t)tp_ * DMOD + d0, h);
        b0 = wmma_bf16(arr_r[s], brn, b0);
        b1 = wmma_bf16(arr_r[s], brp, b1);
        b2 = wmma_bf16(arr_p[s], brn, b2);
        b3 = wmma_bf16(arr_p[s], brp, b3);
      }
      ac_r[sj] = a0; ac_pp[sj] = a1; ac_rp[sj] = a2;
#pragma unroll
      for (int rr = 0; rr < 8; ++rr) {
        int m = rr + 8 * h;
        bd_r[w][sj][m][nn]      = b0[rr];
        bd_r[w][sj][m][16 + nn] = b1[rr];
        bd_p[w][sj][m][nn]      = b2[rr];
        bd_p[w][sj][m][16 + nn] = b3[rr];
      }
    }
    __syncthreads();

    // ---- gather rel-shift diagonal, combine magnitudes, mask ----
    v8f stile[2];
#pragma unroll
    for (int sj = 0; sj < 2; ++sj) {
      const int jt = j0 + 16 * sj;
#pragma unroll
      for (int rr = 0; rr < 8; ++rr) {
        int m = rr + 8 * h;
        int col = nn - m + 16;  // in [1,31]
        float bdr = bd_r[w][sj][m][col];
        float bdp = bd_p[w][sj][m][col];
        float acr = ac_r[sj][rr] - ac_pp[sj][rr];
        float acp = 2.0f * ac_rp[sj][rr];
        float sc = (sqrtf(acr * acr + acp * acp) +
                    sqrtf(bdr * bdr + bdp * bdp)) * scale;
        int ig = iw + m, jg = jt + nn;
        stile[sj][rr] = (jg <= ig) ? sc : -INFINITY;
      }
    }

    // ---- online softmax update + write P (bf16) to LDS ----
    float corr[8];
#pragma unroll
    for (int rr = 0; rr < 8; ++rr) {
      float mt = fmaxf(stile[0][rr], stile[1][rr]);
#pragma unroll
      for (int msk = 8; msk >= 1; msk >>= 1)
        mt = fmaxf(mt, __shfl_xor(mt, msk, 32));
      float mnew = fmaxf(mrow[rr], mt);
      float p0 = __expf(stile[0][rr] - mnew);
      float p1 = __expf(stile[1][rr] - mnew);
      float rs = p0 + p1;
#pragma unroll
      for (int msk = 8; msk >= 1; msk >>= 1)
        rs += __shfl_xor(rs, msk, 32);
      float c = __expf(mrow[rr] - mnew);
      corr[rr] = c;
      mrow[rr] = mnew;
      lrow[rr] = lrow[rr] * c + rs;
      int m = rr + 8 * h;
      pbuf[w][m][nn]      = (__bf16)p0;
      pbuf[w][m][16 + nn] = (__bf16)p1;
    }
#pragma unroll
    for (int t = 0; t < 4; ++t)
#pragma unroll
      for (int rr = 0; rr < 8; ++rr) {
        out_r[t][rr] *= corr[rr];
        out_p[t][rr] *= corr[rr];
      }
    __syncthreads();

    // ---- P @ V  (K = 32), V^T gives contiguous B fragments ----
    v16bf ap = a_frag(&pbuf[w][nn][0], h);
#pragma unroll
    for (int t = 0; t < 4; ++t) {
      v16bf bvr = bT_frag(vr + (size_t)(16 * t + nn) * QLEN + j0, h);
      v16bf bvp = bT_frag(vp + (size_t)(16 * t + nn) * QLEN + j0, h);
      out_r[t] = wmma_bf16(ap, bvr, out_r[t]);
      out_p[t] = wmma_bf16(ap, bvp, out_p[t]);
    }
    __syncthreads();  // protect LDS reuse next iteration
  }

  // ---- epilogue: normalize, write vec (bf16) ----
#pragma unroll
  for (int rr = 0; rr < 8; ++rr) {
    int m = rr + 8 * h;
    size_t row = (size_t)((iw + m) * BSZ + b);
    float inv = 1.0f / lrow[rr];
#pragma unroll
    for (int t = 0; t < 4; ++t) {
      vec_r[row * DMOD + n * DHEAD + 16 * t + nn] = (__bf16)(out_r[t][rr] * inv);
      vec_p[row * DMOD + n * DHEAD + 16 * t + nn] = (__bf16)(out_p[t][rr] * inv);
    }
  }
}

// ---------------------------------------------------------------------------
// In-place LayerNorm over last dim (512), one row per block.
// ---------------------------------------------------------------------------
__launch_bounds__(256)
__global__ void ln_kernel(float* __restrict__ y,
                          const float* __restrict__ gamma,
                          const float* __restrict__ beta) {
  float* p = y + (size_t)blockIdx.x * DMOD;
  int tid = threadIdx.x;
  float x0 = p[tid], x1 = p[tid + 256];
  __shared__ float part[8];
  __shared__ float stat[2];
  float s = x0 + x1;
#pragma unroll
  for (int m = 16; m >= 1; m >>= 1) s += __shfl_xor(s, m, 32);
  if ((tid & 31) == 0) part[tid >> 5] = s;
  __syncthreads();
  if (tid == 0) {
    float t = 0.f;
#pragma unroll
    for (int i = 0; i < 8; ++i) t += part[i];
    stat[0] = t * (1.0f / DMOD);
  }
  __syncthreads();
  float mu = stat[0];
  float d0 = x0 - mu, d1 = x1 - mu;
  float vs = d0 * d0 + d1 * d1;
#pragma unroll
  for (int m = 16; m >= 1; m >>= 1) vs += __shfl_xor(vs, m, 32);
  if ((tid & 31) == 0) part[tid >> 5] = vs;
  __syncthreads();
  if (tid == 0) {
    float t = 0.f;
#pragma unroll
    for (int i = 0; i < 8; ++i) t += part[i];
    stat[1] = rsqrtf(t * (1.0f / DMOD) + 1e-5f);
  }
  __syncthreads();
  float rstd = stat[1];
  p[tid]       = d0 * rstd * gamma[tid]       + beta[tid];
  p[tid + 256] = d1 * rstd * gamma[tid + 256] + beta[tid + 256];
}

// ---------------------------------------------------------------------------
// Host launcher
// ---------------------------------------------------------------------------
extern "C" void kernel_launch(void* const* d_in, const int* in_sizes, int n_in,
                              void* d_out, int out_size, void* d_ws, size_t ws_size,
                              hipStream_t stream) {
  const float* w_real  = (const float*)d_in[0];
  const float* w_phase = (const float*)d_in[1];
  const float* r       = (const float*)d_in[2];
  const float* rwb     = (const float*)d_in[3];
  const float* rrb     = (const float*)d_in[4];
  // d_in[5] attn_mask: causal, computed analytically in-kernel
  const float* Wqkv_r  = (const float*)d_in[6];
  const float* Wqkv_p  = (const float*)d_in[7];
  const float* Wr      = (const float*)d_in[8];
  const float* Wo_r    = (const float*)d_in[9];
  const float* Wo_p    = (const float*)d_in[10];
  const float* gamma   = (const float*)d_in[11];
  const float* beta    = (const float*)d_in[12];
  float* out = (float*)d_out;

  // ---- workspace carve-up (all bf16 elems) ----
  __bf16* p = (__bf16*)d_ws;
  __bf16* hr    = p; p += (size_t)MROWS * HCOLS;   // heads real  [4096][1536]
  __bf16* hp    = p; p += (size_t)MROWS * HCOLS;   // heads phase
  __bf16* rh    = p; p += (size_t)QLEN * DMOD;     // r_head_k    [1024][512]
  __bf16* rb    = p; p += (size_t)QLEN * DMOD;     // r (bf16)
  __bf16* wbr   = p; p += (size_t)MROWS * DMOD;    // w_real bf16 [4096][512]
  __bf16* wbp   = p; p += (size_t)MROWS * DMOD;
  __bf16* qkvrT = p; p += (size_t)HCOLS * DMOD;    // Wqkv_r^T [1536][512]
  __bf16* qkvpT = p; p += (size_t)HCOLS * DMOD;
  __bf16* wrT   = p; p += (size_t)DMOD * DMOD;     // Wr^T
  __bf16* worT  = p; p += (size_t)DMOD * DMOD;     // Wo_r^T
  __bf16* wopT  = p; p += (size_t)DMOD * DMOD;     // Wo_p^T
  __bf16* vtr   = p; p += (size_t)BSZ * NHEAD * DHEAD * QLEN;  // V^T real
  __bf16* vtp   = p; p += (size_t)BSZ * NHEAD * DHEAD * QLEN;  // V^T phase
  __bf16* vecr  = p; p += (size_t)MROWS * DMOD;    // attn out real (bf16)
  __bf16* vecp  = p; p += (size_t)MROWS * DMOD;

  // ---- prep: converts + weight transposes ----
  convert_bf16<<<(MROWS * DMOD / 4 + 255) / 256, 256, 0, stream>>>(w_real, wbr,
                                                                   MROWS * DMOD / 4);
  convert_bf16<<<(MROWS * DMOD / 4 + 255) / 256, 256, 0, stream>>>(w_phase, wbp,
                                                                   MROWS * DMOD / 4);
  convert_bf16<<<(QLEN * DMOD / 4 + 255) / 256, 256, 0, stream>>>(r, rb,
                                                                  QLEN * DMOD / 4);
  transpose_to_bf16<<<dim3(DMOD / 32, HCOLS / 32), 256, 0, stream>>>(Wqkv_r, qkvrT,
                                                                     DMOD, HCOLS);
  transpose_to_bf16<<<dim3(DMOD / 32, HCOLS / 32), 256, 0, stream>>>(Wqkv_p, qkvpT,
                                                                     DMOD, HCOLS);
  transpose_to_bf16<<<dim3(DMOD / 32, DMOD / 32), 256, 0, stream>>>(Wr, wrT,
                                                                    DMOD, DMOD);
  transpose_to_bf16<<<dim3(DMOD / 32, DMOD / 32), 256, 0, stream>>>(Wo_r, worT,
                                                                    DMOD, DMOD);
  transpose_to_bf16<<<dim3(DMOD / 32, DMOD / 32), 256, 0, stream>>>(Wo_p, wopT,
                                                                    DMOD, DMOD);

  // ---- projections (WMMA) ----
  {
    int waves = (MROWS / 32) * (HCOLS / 64);
    gemm_bf16<<<(waves + 3) / 4, 128, 0, stream>>>(wbr, qkvrT, hr,
                                                   MROWS, HCOLS, DMOD);
    gemm_bf16<<<(waves + 3) / 4, 128, 0, stream>>>(wbp, qkvpT, hp,
                                                   MROWS, HCOLS, DMOD);
    int waves_r = (QLEN / 32) * (DMOD / 64);
    gemm_bf16<<<(waves_r + 3) / 4, 128, 0, stream>>>(rb, wrT, rh,
                                                     QLEN, DMOD, DMOD);
  }

  // ---- V repack to per-head transposed layout (async-to-LDS path) ----
  repack_v<<<dim3(QLEN / 64, BSZ * NHEAD), 256, 0, stream>>>(hr, vtr);
  repack_v<<<dim3(QLEN / 64, BSZ * NHEAD), 256, 0, stream>>>(hp, vtp);

  // ---- fused relative attention (flash softmax, WMMA) ----
  attn_kernel<<<dim3(QLEN / 64, BSZ * NHEAD), 128, 0, stream>>>(
      hr, hp, rh, vtr, vtp, rwb, rrb, vecr, vecp);

  // ---- output projection + residual (WMMA), then in-place LayerNorm ----
  {
    int waves = (MROWS / 32) * (DMOD / 64);
    gemm_resid<<<(waves + 3) / 4, 128, 0, stream>>>(vecr, worT, w_real, out,
                                                    MROWS, DMOD, DMOD);
    gemm_resid<<<(waves + 3) / 4, 128, 0, stream>>>(
        vecp, wopT, w_phase, out + (size_t)MROWS * DMOD, MROWS, DMOD, DMOD);
  }
  ln_kernel<<<MROWS, 256, 0, stream>>>(out, gamma, beta);
  ln_kernel<<<MROWS, 256, 0, stream>>>(out + (size_t)MROWS * DMOD, gamma, beta);
}
